// ColumnSelfAttention_81209241632884
// MI455X (gfx1250) — matrix operand order repaired and checked by reading, not compiled, not run on hardware
//
#include <hip/hip_runtime.h>
#include <hip/hip_bf16.h>

// ---------- types ----------
typedef __attribute__((ext_vector_type(16))) _Float16 v16h;
typedef __attribute__((ext_vector_type(8)))  _Float16 h8_t;
typedef __attribute__((ext_vector_type(4)))  _Float16 h4_t;
typedef __attribute__((ext_vector_type(8)))  float    v8f;

union V16 { v16h v; h8_t h[2]; };

__device__ __forceinline__ v8f wmma_f16(v16h a, v16h b, v8f c) {
  return __builtin_amdgcn_wmma_f32_16x16x32_f16(
      /*neg_a=*/false, a, /*neg_b=*/false, b,
      /*c_mod=*/(short)0, c, /*reuse_a=*/false, /*reuse_b=*/false);
}

// ---------- fp32 -> fp16 convert ----------
__global__ void cvt_f32_f16(const float* __restrict__ in,
                            _Float16* __restrict__ out, int n) {
  int i = (blockIdx.x * blockDim.x + threadIdx.x) * 4;
  if (i + 3 < n) {
    float4 f = *(const float4*)(in + i);
    h4_t h;
    h.x = (_Float16)f.x; h.y = (_Float16)f.y;
    h.z = (_Float16)f.z; h.w = (_Float16)f.w;
    *(h4_t*)(out + i) = h;
  }
}

// ---------- WMMA GEMM: Y = scale * (A @ W^T + bias) ----------
// A: [M=65536, 768] f16 (row m = i*C + c), W: [768,768] f16 (row-major, K contig)
// MODE 0: f16 out, attention layout out[((c*H+h)*R + i)*64 + d], n = h*64+d
// MODE 1: f32 out, flat out[m*768 + n]
template <int MODE>
__global__ void gemm_xwT(const _Float16* __restrict__ A,
                         const _Float16* __restrict__ W,
                         const float* __restrict__ bias,
                         void* __restrict__ out, float scale) {
  const int K = 768;
  const int lane   = threadIdx.x & 31;
  const int wave   = threadIdx.x >> 5;      // 0..7
  const int wm     = wave >> 1;             // 0..3
  const int wn     = wave & 1;              // 0..1
  const int mBase  = blockIdx.x * 128 + wm * 32;
  const int nBase  = blockIdx.y * 64  + wn * 32;
  const int halfId = lane >> 4;
  const int l16    = lane & 15;

  v8f acc[2][2] = {};
  for (int kk = 0; kk < K; kk += 32) {
    V16 a[2], b[2];
#pragma unroll
    for (int im = 0; im < 2; ++im) {
      const _Float16* ap = A + (size_t)(mBase + im * 16 + l16) * K + kk + halfId * 8;
      a[im].h[0] = *(const h8_t*)(ap);
      a[im].h[1] = *(const h8_t*)(ap + 16);
    }
#pragma unroll
    for (int jn = 0; jn < 2; ++jn) {
      const _Float16* bp = W + (size_t)(nBase + jn * 16 + l16) * K + kk + halfId * 8;
      b[jn].h[0] = *(const h8_t*)(bp);
      b[jn].h[1] = *(const h8_t*)(bp + 16);
    }
#pragma unroll
    for (int im = 0; im < 2; ++im)
#pragma unroll
      for (int jn = 0; jn < 2; ++jn)
        acc[im][jn] = wmma_f16(a[im].v, b[jn].v, acc[im][jn]);
  }

#pragma unroll
  for (int im = 0; im < 2; ++im) {
#pragma unroll
    for (int jn = 0; jn < 2; ++jn) {
      int n = nBase + jn * 16 + l16;
      float bv = bias[n];
#pragma unroll
      for (int r = 0; r < 8; ++r) {
        int m = mBase + im * 16 + r + halfId * 8;
        float y = (acc[im][jn][r] + bv) * scale;
        if (MODE == 0) {
          int i = m >> 9, c = m & 511;        // C = 512
          int h = n >> 6, d = n & 63;         // D = 64
          ((_Float16*)out)[(((size_t)c * 12 + h) * 128 + i) * 64 + d] = (_Float16)y;
        } else {
          ((float*)out)[(size_t)m * 768 + n] = y;
        }
      }
    }
  }
}

// ---------- attention per (c,h): S = QK^T, mask, softmax, ctx = P V ----------
#define PSTR 136  // padded LDS row stride (halves)

__global__ void attn_kernel(const _Float16* __restrict__ Q,   // [C,H,R,D]
                            const _Float16* __restrict__ Km,  // [C,H,R,D]
                            const _Float16* __restrict__ V,   // [C,H,R,D]
                            const unsigned char* __restrict__ pmask, // [1,R,C]
                            float* __restrict__ attnOut,      // [H,C,R,R]
                            _Float16* __restrict__ ctx) {     // [M,768]
  const int R_ = 128, D_ = 64, C_ = 512, H_ = 12;
  const int h = blockIdx.x % H_;
  const int c = blockIdx.x / H_;
  const size_t tileOff = (size_t)blockIdx.x * (R_ * D_); // (c*H+h)*R*D
  const _Float16* q = Q + tileOff;
  const _Float16* k = Km + tileOff;
  const _Float16* v = V + tileOff;

  __shared__ _Float16 VT[64 * PSTR];    // VT[d*PSTR + j]
  __shared__ _Float16 P[128 * PSTR];    // P[i*PSTR + j]

  const int tid = threadIdx.x;
  const int lane = tid & 31;
  const int wave = tid >> 5;
  const int halfId = lane >> 4;
  const int l16 = lane & 15;
  const int iBase = wave * 16;

  // stage transposed V into LDS
  for (int e = tid; e < R_ * D_; e += 256) {
    int j = e >> 6, d = e & 63;
    VT[d * PSTR + j] = v[e];
  }
  __syncthreads();

  // S strip [16 x 128] for this wave
  v8f s[8] = {};
  for (int kk = 0; kk < D_; kk += 32) {
    V16 a;
    const _Float16* ap = q + (size_t)(iBase + l16) * D_ + kk + halfId * 8;
    a.h[0] = *(const h8_t*)(ap);
    a.h[1] = *(const h8_t*)(ap + 16);
#pragma unroll
    for (int t = 0; t < 8; ++t) {
      V16 b;
      const _Float16* bp = k + (size_t)(t * 16 + l16) * D_ + kk + halfId * 8;
      b.h[0] = *(const h8_t*)(bp);
      b.h[1] = *(const h8_t*)(bp + 16);
      s[t] = wmma_f16(a.v, b.v, s[t]);
    }
  }

  // padding mask: attn[..., j] = -10000 where pmask[0, j, c]
#pragma unroll
  for (int t = 0; t < 8; ++t) {
    int j = t * 16 + l16;
    bool msk = pmask[(size_t)j * C_ + c] != 0;
#pragma unroll
    for (int r = 0; r < 8; ++r)
      s[t][r] = msk ? -10000.0f : s[t][r];
  }

  // row softmax: row's 128 values live on 16 lanes (same lane>>4) x 8 tiles
  float rmax[8], rsum[8];
#pragma unroll
  for (int r = 0; r < 8; ++r) {
    float mx = s[0][r];
#pragma unroll
    for (int t = 1; t < 8; ++t) mx = fmaxf(mx, s[t][r]);
#pragma unroll
    for (int o = 1; o < 16; o <<= 1) mx = fmaxf(mx, __shfl_xor(mx, o, 32));
    rmax[r] = mx;
  }
#pragma unroll
  for (int r = 0; r < 8; ++r) {
    float sm = 0.0f;
#pragma unroll
    for (int t = 0; t < 8; ++t) {
      float p = __expf(s[t][r] - rmax[r]);
      s[t][r] = p;
      sm += p;
    }
#pragma unroll
    for (int o = 1; o < 16; o <<= 1) sm += __shfl_xor(sm, o, 32);
    rsum[r] = sm;
  }

  // normalize; emit probs to d_out (f32) and P (f16) to LDS
  float* aBase = attnOut + ((size_t)h * C_ + c) * R_ * R_;
#pragma unroll
  for (int r = 0; r < 8; ++r) {
    float inv = 1.0f / rsum[r];
    int i = iBase + r + halfId * 8;
#pragma unroll
    for (int t = 0; t < 8; ++t) {
      float p = s[t][r] * inv;
      int j = t * 16 + l16;
      aBase[(size_t)i * R_ + j] = p;
      P[i * PSTR + j] = (_Float16)p;
    }
  }
  __syncthreads();

  // ctx strip [16 x 64] = P(strip) @ V
  v8f cacc[4] = {};
  for (int jk = 0; jk < R_; jk += 32) {
    V16 a;
    const _Float16* ap = &P[(iBase + l16) * PSTR + jk + halfId * 8];
    a.h[0] = *(const h8_t*)(ap);
    a.h[1] = *(const h8_t*)(ap + 16);
#pragma unroll
    for (int t = 0; t < 4; ++t) {
      V16 b;
      const _Float16* bp = &VT[(t * 16 + l16) * PSTR + jk + halfId * 8];
      b.h[0] = *(const h8_t*)(bp);
      b.h[1] = *(const h8_t*)(bp + 16);
      cacc[t] = wmma_f16(a.v, b.v, cacc[t]);
    }
  }

  // scatter ctx (f16) into flat [M,768]: ctx[(i*C + c)*768 + h*64 + d]
#pragma unroll
  for (int t = 0; t < 4; ++t) {
    int d = t * 16 + l16;
    int e = h * 64 + d;
#pragma unroll
    for (int r = 0; r < 8; ++r) {
      int i = iBase + r + halfId * 8;
      ctx[((size_t)(i * C_ + c)) * 768 + e] = (_Float16)cacc[t][r];
    }
  }
}

// ---------- host launch ----------
extern "C" void kernel_launch(void* const* d_in, const int* in_sizes, int n_in,
                              void* d_out, int out_size, void* d_ws, size_t ws_size,
                              hipStream_t stream) {
  const float* x  = (const float*)d_in[0];
  const float* Wq = (const float*)d_in[1];
  const float* bq = (const float*)d_in[2];
  const float* Wk = (const float*)d_in[3];
  const float* bk = (const float*)d_in[4];
  const float* Wv = (const float*)d_in[5];
  const float* bv = (const float*)d_in[6];
  const float* Wo = (const float*)d_in[7];
  const float* bo = (const float*)d_in[8];
  const unsigned char* pmask = (const unsigned char*)d_in[9];

  const int R = 128, C = 512, E = 768, H = 12;
  const size_t M   = (size_t)R * C;   // 65536
  const size_t ME  = M * E;           // 50,331,648 elements
  const size_t EE  = (size_t)E * E;   // 589,824

  char* ws = (char*)d_ws;
  _Float16* xh  = (_Float16*)ws;                     // [M,E] f16 (reused as ctx)
  _Float16* qws = (_Float16*)(ws + ME * 2 * 1);      // [C,H,R,D]
  _Float16* kws = (_Float16*)(ws + ME * 2 * 2);
  _Float16* vws = (_Float16*)(ws + ME * 2 * 3);
  _Float16* whq = (_Float16*)(ws + ME * 2 * 4);
  _Float16* whk = whq + EE;
  _Float16* whv = whq + EE * 2;
  _Float16* who = whq + EE * 3;
  _Float16* ctxh = xh;                               // alias: xh dead after QKV

  // 1) converts
  cvt_f32_f16<<<dim3((unsigned)(ME / 1024)), 256, 0, stream>>>(x,  xh,  (int)ME);
  cvt_f32_f16<<<dim3((unsigned)(EE / 1024)), 256, 0, stream>>>(Wq, whq, (int)EE);
  cvt_f32_f16<<<dim3((unsigned)(EE / 1024)), 256, 0, stream>>>(Wk, whk, (int)EE);
  cvt_f32_f16<<<dim3((unsigned)(EE / 1024)), 256, 0, stream>>>(Wv, whv, (int)EE);
  cvt_f32_f16<<<dim3((unsigned)(EE / 1024)), 256, 0, stream>>>(Wo, who, (int)EE);

  // 2) projections (q scaled by D^-0.5 = 0.125)
  dim3 ggrid((unsigned)(M / 128), (unsigned)(E / 64));
  gemm_xwT<0><<<ggrid, 256, 0, stream>>>(xh, whq, bq, (void*)qws, 0.125f);
  gemm_xwT<0><<<ggrid, 256, 0, stream>>>(xh, whk, bk, (void*)kws, 1.0f);
  gemm_xwT<0><<<ggrid, 256, 0, stream>>>(xh, whv, bv, (void*)vws, 1.0f);

  // 3) attention per (c,h); probs go after `out` in d_out
  float* attnOut = (float*)d_out + ME;
  attn_kernel<<<dim3((unsigned)(C * H)), 256, 0, stream>>>(
      qws, kws, vws, pmask, attnOut, ctxh);

  // 4) output projection -> f32 flat into d_out
  gemm_xwT<1><<<ggrid, 256, 0, stream>>>(ctxh, who, bo, d_out, 1.0f);
}
